// RNNCRF_28071906247322
// MI455X (gfx1250) — matrix operand order repaired and checked by reading, not compiled, not run on hardware
//
#include <hip/hip_runtime.h>
#include <cstdint>
#include <cstddef>

#define E   512
#define H   512
#define TT  21
#define BB  64
#define LL  512
#define G4  2048            // 4*H
#define MM  (LL*BB)         // 32768

typedef __attribute__((ext_vector_type(16))) __bf16 bf16x16;
typedef __attribute__((ext_vector_type(8)))  __bf16 bf16x8;
typedef __attribute__((ext_vector_type(4)))  __bf16 bf16x4;
typedef __attribute__((ext_vector_type(8)))  float  f32x8;
typedef unsigned int v4u __attribute__((ext_vector_type(4)));
typedef int          v8i __attribute__((ext_vector_type(8)));
typedef int          v4i __attribute__((ext_vector_type(4)));

// ---- WMMA fragment loaders (bf16 16x16x32, per CDNA5 VGPR layouts) ----
__device__ __forceinline__ bf16x16 load_frag_a(const __bf16* rowBase, int half) {
  bf16x8 lo = *(const bf16x8*)(rowBase + (half ? 8 : 0));
  bf16x8 hi = *(const bf16x8*)(rowBase + (half ? 24 : 16));
  bf16x16 r;
#pragma unroll
  for (int i = 0; i < 8; ++i) { r[i] = lo[i]; r[i + 8] = hi[i]; }
  return r;
}
__device__ __forceinline__ bf16x16 load_frag_b(const __bf16* colBase, int half) {
  bf16x8 lo = *(const bf16x8*)(colBase + half * 8);
  bf16x8 hi = *(const bf16x8*)(colBase + 16 + half * 8);
  bf16x16 r;
#pragma unroll
  for (int i = 0; i < 8; ++i) { r[i] = lo[i]; r[i + 8] = hi[i]; }
  return r;
}

__device__ __forceinline__ float sigm(float x)   { return 1.f / (1.f + __expf(-x)); }
__device__ __forceinline__ float tanh_f(float x) { return 2.f / (1.f + __expf(-2.f * x)) - 1.f; }

// ---- Tensor Data Mover: 2D tile (rows x rowlen bf16), global->LDS, with LDS
// padding of 4 DWORDs (8 bf16) after every 256 DWORDs (512 bf16) so a 512-elem
// row lands at LDS stride 520 (bank-conflict-free fragment loads). ----
__device__ __forceinline__ void tdm_load_2d(unsigned lds_byte, const void* gaddr,
                                            unsigned rows, unsigned rowlen,
                                            unsigned stride_elems) {
  const unsigned long long ga = (unsigned long long)gaddr;
  v4u g0;
  g0[0] = 1u;                                             // count=1 (user D#)
  g0[1] = lds_byte;                                       // lds_addr
  g0[2] = (unsigned)(ga & 0xffffffffu);                   // global_addr[31:0]
  g0[3] = (unsigned)((ga >> 32) & 0x01ffffffu) | (2u << 30);  // addr[56:32] | type=2
  v8i g1;
  g1[0] = (int)((1u << 16)        // data_size = 2 bytes
              | (1u << 20)        // pad_enable
              | (7u << 22)        // pad_interval: 256 DWORDs
              | (3u << 25));      // pad_amount:   4 DWORDs
  g1[1] = (int)((rowlen & 0xffffu) << 16);                         // tensor_dim0 lo
  g1[2] = (int)(((rowlen >> 16) & 0xffffu) | ((rows & 0xffffu) << 16)); // dim0 hi | dim1 lo
  g1[3] = (int)(((rows >> 16) & 0xffffu) | ((rowlen & 0xffffu) << 16)); // dim1 hi | tile_dim0
  g1[4] = (int)(rows & 0xffffu);                                   // tile_dim1 (tile_dim2=0)
  g1[5] = (int)stride_elems;                                       // tensor_dim0_stride lo
  g1[6] = 0;
  g1[7] = 0;
  v4i z = {0, 0, 0, 0};
#if __clang_major__ >= 23
  v8i z8 = {0, 0, 0, 0, 0, 0, 0, 0};
  __builtin_amdgcn_tensor_load_to_lds(g0, g1, z, z, z8, 0);
#else
  __builtin_amdgcn_tensor_load_to_lds(g0, g1, z, z, 0);
#endif
}

// ---------------- init: zero h0 double-buffers, sync counters, output ----------------
__global__ void k_init(__bf16* hstate, unsigned* sync, float* out) {
  int i = blockIdx.x * blockDim.x + threadIdx.x;
  if (i < 2 * 2 * BB * H) hstate[i] = (__bf16)0.f;
  if (i == 0) { sync[0] = 0u; sync[1] = 0u; out[0] = 0.f; }
}

// ---------------- f32 -> bf16 weight conversion ----------------
__global__ void k_cvt(const float* __restrict__ src, __bf16* __restrict__ dst, int n) {
  int i = blockIdx.x * blockDim.x + threadIdx.x;
  if (i < n) dst[i] = (__bf16)src[i];
}

// ---------------- fused embedding gather + input projection GEMM (bf16 WMMA) ----------
__global__ void __launch_bounds__(256)
k_xproj(const int* __restrict__ x, const float* __restrict__ emb,
        const __bf16* __restrict__ Wih,
        const float* __restrict__ bih_f, const float* __restrict__ bhh_f,
        const float* __restrict__ bih_b, const float* __restrict__ bhh_b,
        __bf16* __restrict__ xg) {
  __shared__ __bf16 As[128 * 40];
  __shared__ __bf16 Bs[128 * 40];
  __shared__ int tok[128];

  const int tid   = threadIdx.x;
  const int dir   = blockIdx.z;
  const int tileN = blockIdx.x * 128;
  const int tileM = blockIdx.y * 128;
  const __bf16* W   = Wih + (size_t)dir * G4 * E;
  __bf16*       out = xg  + (size_t)dir * (size_t)MM * G4;
  const float*  bi  = dir ? bih_b : bih_f;
  const float*  bh  = dir ? bhh_b : bhh_f;

  if (tid < 128) {
    int m = tileM + tid;                       // m = l*B + b
    tok[tid] = x[(m & 63) * LL + (m >> 6)];    // x is (B,L)
  }
  __syncthreads();

  const int w = tid >> 5, lane = tid & 31, half = lane >> 4, lr = lane & 15;

  f32x8 acc[8] = {};
#pragma unroll 2
  for (int kc = 0; kc < E / 32; ++kc) {
    const int k0 = kc * 32;
#pragma unroll
    for (int it = 0; it < 4; ++it) {
      int g = tid + it * 256;
      int row = g >> 3, k4 = (g & 7) * 4;
      const float4 va = *(const float4*)(emb + (size_t)tok[row] * E + k0 + k4);
      bf16x4 pa; pa[0] = (__bf16)va.x; pa[1] = (__bf16)va.y;
      pa[2] = (__bf16)va.z; pa[3] = (__bf16)va.w;
      *(bf16x4*)(As + row * 40 + k4) = pa;
    }
#pragma unroll
    for (int it = 0; it < 2; ++it) {
      int g = tid + it * 256;
      int row = g >> 2, k8 = (g & 3) * 8;
      *(bf16x8*)(Bs + row * 40 + k8) =
          *(const bf16x8*)(W + (size_t)(tileN + row) * E + k0 + k8);
    }
    __syncthreads();

    bf16x16 bfg = load_frag_b(Bs + (w * 16 + lr) * 40, half);
#pragma unroll
    for (int mi = 0; mi < 8; ++mi) {
      bf16x16 afg = load_frag_a(As + (mi * 16 + lr) * 40, half);
      acc[mi] = __builtin_amdgcn_wmma_f32_16x16x32_bf16(
          false, afg, false, bfg, (short)0, acc[mi], false, false);
    }
    __syncthreads();
  }

  const int ncol = tileN + w * 16 + lr;
  const float bias = bi[ncol] + bh[ncol];
#pragma unroll
  for (int mi = 0; mi < 8; ++mi)
#pragma unroll
    for (int r = 0; r < 8; ++r) {
      int row = tileM + mi * 16 + (half ? r + 8 : r);
      out[(size_t)row * G4 + ncol] = (__bf16)(acc[mi][r] + bias);
    }
}

// ---------------- persistent cooperative LSTM (both directions) ----------------
__global__ void __launch_bounds__(256)
k_lstm(const __bf16* __restrict__ xg, const __bf16* __restrict__ Whh,
       __bf16* __restrict__ hstate, float* __restrict__ hf, float* __restrict__ hb,
       unsigned* __restrict__ sync) {
  extern __shared__ char smem[];
  __bf16* Ws   = (__bf16*)smem;                            // [128][520] 130KB, resident
  __bf16* hsld = (__bf16*)(smem + 133120);                 // [64][520]   65KB
  float*  gs   = (float*)(smem + 133120 + 66560);          // [64][132]   33KB
  float*  cs   = (float*)(smem + 133120 + 66560 + 33792);  // [64][32]     8KB

  const int tid = threadIdx.x;
  const int dir = blockIdx.y;
  const int wg  = blockIdx.x;
  const int j0  = wg * 32;
  const __bf16* xgd = xg  + (size_t)dir * (size_t)MM * G4;
  const __bf16* Wd  = Whh + (size_t)dir * G4 * H;
  __bf16* hbuf = hstate + (size_t)dir * 2 * BB * H;        // [2][64][512]
  float*  hs   = dir ? hb : hf;
  unsigned* syn = sync + dir;

  // TDM-stage the resident W_hh slice: 4 gate blocks of 32 consecutive rows.
  if (tid < 32) {
#pragma unroll
    for (int g = 0; g < 4; ++g) {
      int ng0 = (g << 9) + j0;
      tdm_load_2d((unsigned)(uintptr_t)Ws + (unsigned)(g * 32 * 1040),
                  Wd + (size_t)ng0 * H, 32u, 512u, 512u);
    }
    __builtin_amdgcn_s_wait_tensorcnt(0);
  }
  for (int u = tid; u < BB * 32; u += 256) cs[u] = 0.f;
  __syncthreads();

  const int w = tid >> 5, lane = tid & 31, half = lane >> 4, lr = lane & 15;
  const int cme = w * 16 + lr;                       // this lane's local column
  const int ngme = ((cme >> 5) << 9) + j0 + (cme & 31);

  for (int s = 0; s < LL; ++s) {
    const int l  = dir ? (LL - 1 - s) : s;
    const int rb = s & 1;
    if (s > 0) {
      if (tid == 0) {
        while (__hip_atomic_load(syn, __ATOMIC_ACQUIRE, __HIP_MEMORY_SCOPE_AGENT) <
               16u * (unsigned)s)
          __builtin_amdgcn_s_sleep(2);
      }
      __syncthreads();
    }
    // TDM-stage h_prev (64x512 bf16) into LDS with hardware padding to stride 520
    if (tid < 32) {
      tdm_load_2d((unsigned)(uintptr_t)hsld, hbuf + rb * BB * H, 64u, 512u, 512u);
      __builtin_amdgcn_s_wait_tensorcnt(0);
    }
    // speculative prefetch of next step's xg slice while this step computes
    {
      int ln = dir ? (l - 1) : (l + 1);
      if (ln >= 0 && ln < LL) {
        const __bf16* p =
            xgd + (size_t)(ln * BB + (tid & 63)) * G4 + j0 + ((tid >> 6) << 3);
        __builtin_prefetch((const void*)p, 0, 1);
      }
    }
    __syncthreads();

    // gates = h_prev @ Whh_sliceT ; wave w handles 16 of the WG's 128 columns
    f32x8 acc[4] = {};
#pragma unroll 2
    for (int kc = 0; kc < H / 32; ++kc) {
      bf16x16 bfg = load_frag_b(Ws + cme * 520 + kc * 32, half);
#pragma unroll
      for (int mi = 0; mi < 4; ++mi) {
        bf16x16 afg = load_frag_a(hsld + (mi * 16 + lr) * 520 + kc * 32, half);
        acc[mi] = __builtin_amdgcn_wmma_f32_16x16x32_bf16(
            false, afg, false, bfg, (short)0, acc[mi], false, false);
      }
    }
    {
      const __bf16* xrow = xgd + (size_t)(l * BB) * G4 + ngme;
#pragma unroll
      for (int mi = 0; mi < 4; ++mi)
#pragma unroll
        for (int r = 0; r < 8; ++r) {
          int b = mi * 16 + (half ? r + 8 : r);
          gs[b * 132 + cme] = acc[mi][r] + (float)xrow[(size_t)b * G4];
        }
    }
    __syncthreads();

    // pointwise LSTM cell update for our 64x32 hidden slice
    __bf16* hw = hbuf + (1 - rb) * BB * H;
    for (int u = tid; u < BB * 32; u += 256) {
      int b = u >> 5, jj = u & 31;
      float gi = gs[b * 132 + jj];
      float gf = gs[b * 132 + 32 + jj];
      float gg = gs[b * 132 + 64 + jj];
      float go = gs[b * 132 + 96 + jj];
      float cn = sigm(gf) * cs[u] + sigm(gi) * tanh_f(gg);
      float hn = sigm(go) * tanh_f(cn);
      cs[u] = cn;
      hs[((size_t)l * BB + b) * H + j0 + jj] = hn;
      hw[b * H + j0 + jj] = (__bf16)hn;
    }
    __threadfence();
    __syncthreads();
    if (tid == 0)
      __hip_atomic_fetch_add(syn, 1u, __ATOMIC_RELEASE, __HIP_MEMORY_SCOPE_AGENT);
  }
}

// ---------------- emission: em = [hf,hb] @ W_lin^T + b_lin (N=21, VALU) ------------
__global__ void __launch_bounds__(256)
k_emis(const float* __restrict__ hf, const float* __restrict__ hb,
       const float* __restrict__ Wlin, const float* __restrict__ blin,
       float* __restrict__ em) {
  extern __shared__ float Wl[];  // 21*1024 f32
  const int tid = threadIdx.x;
  for (int e = tid; e < TT * 1024; e += 256) Wl[e] = Wlin[e];
  __syncthreads();
  const int lane = tid & 31;
  const int wid  = blockIdx.x * 8 + (tid >> 5);
  for (int m = wid; m < MM; m += 2048) {
    float xv[32];
#pragma unroll
    for (int q = 0; q < 16; ++q) xv[q]      = hf[(size_t)m * H + lane + q * 32];
#pragma unroll
    for (int q = 0; q < 16; ++q) xv[16 + q] = hb[(size_t)m * H + lane + q * 32];
    for (int t = 0; t < TT; ++t) {
      float p = 0.f;
#pragma unroll
      for (int q = 0; q < 32; ++q) p += xv[q] * Wl[t * 1024 + lane + q * 32];
#pragma unroll
      for (int off = 16; off >= 1; off >>= 1) p += __shfl_xor(p, off, 32);
      if (lane == 0) em[(size_t)m * TT + t] = p + blin[t];
    }
  }
}

// ---------------- CRF NLL: one wave per batch element ----------------
__global__ void __launch_bounds__(32)
k_crf(const float* __restrict__ em, const int* __restrict__ y,
      const int* __restrict__ seq_lens, const float* __restrict__ start_tr,
      const float* __restrict__ end_tr, const float* __restrict__ trans,
      float* __restrict__ out) {
  __shared__ float tr[TT * TT];
  const int b = blockIdx.x, lane = threadIdx.x;
  for (int e = lane; e < TT * TT; e += 32) tr[e] = trans[e];
  __syncthreads();
  const int seq = seq_lens[b];
  const int ln  = lane < TT ? lane : 0;
  const float NEG = -3.0e38f;

  float alpha = (lane < TT) ? start_tr[lane] + em[(size_t)b * TT + lane] : NEG;
  for (int l = 1; l < LL; ++l) {
    if (l < seq) {
      float e = (lane < TT) ? em[((size_t)l * BB + b) * TT + lane] : 0.f;
      float mx = NEG;
#pragma unroll
      for (int i = 0; i < TT; ++i) {
        float ai = __shfl(alpha, i, 32);
        mx = fmaxf(mx, ai + tr[i * TT + ln]);
      }
      float sm = 0.f;
#pragma unroll
      for (int i = 0; i < TT; ++i) {
        float ai = __shfl(alpha, i, 32);
        sm += __expf(ai + tr[i * TT + ln] - mx);
      }
      float nxt = mx + __logf(sm) + e;
      alpha = (lane < TT) ? nxt : NEG;
    }
  }
  float v = (lane < TT) ? alpha + end_tr[lane] : NEG;
  float mx = v;
#pragma unroll
  for (int off = 16; off >= 1; off >>= 1) mx = fmaxf(mx, __shfl_xor(mx, off, 32));
  float sm = (lane < TT) ? __expf(v - mx) : 0.f;
#pragma unroll
  for (int off = 16; off >= 1; off >>= 1) sm += __shfl_xor(sm, off, 32);
  float logZ = mx + __logf(sm);
  float part = 0.f;
  for (int l = lane; l < LL; l += 32) {
    int tg = y[b * LL + l];
    if (l == 0)       part += start_tr[tg] + em[(size_t)b * TT + tg];
    else if (l < seq) part += tr[y[b * LL + l - 1] * TT + tg] +
                              em[((size_t)l * BB + b) * TT + tg];
  }
#pragma unroll
  for (int off = 16; off >= 1; off >>= 1) part += __shfl_xor(part, off, 32);
  if (lane == 0) {
    float num = part + end_tr[y[b * LL + seq - 1]];
    atomicAdd(out, -(num - logZ));
  }
}

extern "C" void kernel_launch(void* const* d_in, const int* in_sizes, int n_in,
                              void* d_out, int out_size, void* d_ws, size_t ws_size,
                              hipStream_t stream) {
  const int*   x     = (const int*)d_in[0];
  const int*   y     = (const int*)d_in[1];
  const int*   seqs  = (const int*)d_in[2];
  const float* emb   = (const float*)d_in[3];
  const float* Wih_f = (const float*)d_in[4];
  const float* Whh_f = (const float*)d_in[5];
  const float* bih_f = (const float*)d_in[6];
  const float* bhh_f = (const float*)d_in[7];
  const float* Wih_b = (const float*)d_in[8];
  const float* Whh_b = (const float*)d_in[9];
  const float* bih_b = (const float*)d_in[10];
  const float* bhh_b = (const float*)d_in[11];
  const float* Wlin  = (const float*)d_in[12];
  const float* blin  = (const float*)d_in[13];
  const float* st_tr = (const float*)d_in[14];
  const float* en_tr = (const float*)d_in[15];
  const float* trans = (const float*)d_in[16];

  char* ws = (char*)d_ws;
  size_t off = 0;
  auto take = [&](size_t bytes) {
    char* p = ws + off;
    off = (off + bytes + 255) & ~(size_t)255;
    return p;
  };
  __bf16* xg     = (__bf16*)take((size_t)2 * MM * G4 * 2);   // 256 MB
  __bf16* Wih_bf = (__bf16*)take((size_t)2 * G4 * E * 2);    // 4 MB
  __bf16* Whh_bf = (__bf16*)take((size_t)2 * G4 * H * 2);    // 4 MB
  __bf16* hstate = (__bf16*)take((size_t)2 * 2 * BB * H * 2);
  float*  hf     = (float*)take((size_t)MM * H * 4);         // 64 MB
  float*  hb     = (float*)take((size_t)MM * H * 4);         // 64 MB
  float*  em     = (float*)take((size_t)MM * TT * 4);        // 2.75 MB
  unsigned* sync = (unsigned*)take(256);

  k_init<<<512, 256, 0, stream>>>(hstate, sync, (float*)d_out);
  k_cvt<<<4096, 256, 0, stream>>>(Wih_f, Wih_bf, G4 * E);
  k_cvt<<<4096, 256, 0, stream>>>(Wih_b, Wih_bf + (size_t)G4 * E, G4 * E);
  k_cvt<<<4096, 256, 0, stream>>>(Whh_f, Whh_bf, G4 * H);
  k_cvt<<<4096, 256, 0, stream>>>(Whh_b, Whh_bf + (size_t)G4 * H, G4 * H);
  k_xproj<<<dim3(16, 256, 2), 256, 0, stream>>>(x, emb, Wih_bf, bih_f, bhh_f,
                                                bih_b, bhh_b, xg);
  const size_t lstmLds = 133120 + 66560 + 33792 + 8192;  // 241,664 B < 320 KB/WGP
  k_lstm<<<dim3(16, 2), 256, lstmLds, stream>>>(xg, Whh_bf, hstate, hf, hb, sync);
  k_emis<<<256, 256, (size_t)TT * 1024 * 4, stream>>>(hf, hb, Wlin, blin, em);
  k_crf<<<BB, 32, 0, stream>>>(em, y, seqs, st_tr, en_tr, trans, (float*)d_out);
}
